// MaskedAdaptiveHypergraphGenerator_9242769621320
// MI455X (gfx1250) — matrix operand order repaired and checked by reading, not compiled, not run on hardware
//
#include <hip/hip_runtime.h>

#define SEQ     16384
#define DMODEL  128
#define HYPER   256
#define BATCH   32
#define KTOP    8
#define THRESH  0.001f
#define LDSTR   276   // padded row stride (floats): 16B-aligned, de-conflicted banks

typedef __attribute__((ext_vector_type(16))) _Float16 v16h;
typedef __attribute__((ext_vector_type(8)))  float    v8f;
typedef __attribute__((ext_vector_type(4)))  float    v4f;

__global__ __launch_bounds__(256)
void hypergraph_fused_kernel(const float* __restrict__ mask,
                             const float* __restrict__ node_embeds,
                             const float* __restrict__ hyper_embeds,
                             float* __restrict__ out)
{
    __shared__ float simS[16][LDSTR];
    __shared__ float pen[16];

    const int tid = threadIdx.x;
    const int m0  = blockIdx.x * 16;

    // ---- Phase 0: mask penalty = mean over batch, one lane per row ----
    if (tid < 16) {
        float s = 0.0f;
        const float* mp = mask + (m0 + tid);
        #pragma unroll 8
        for (int b = 0; b < BATCH; ++b) s += mp[(size_t)b * SEQ];
        pen[tid] = s * (1.0f / (float)BATCH);
    }

    // ---- Phase 1: sim[16x256] = node[m0:m0+16,:] * hyper^T via f16 WMMA ----
    {
        const int wave = tid >> 5;        // 0..7, each wave -> 2 N-tiles
        const int lane = tid & 31;
        const int half = lane >> 4;       // 0 or 1
        const int l16  = lane & 15;

        // A fragments (16x32 f16), ISA layout: half==0 -> K {0..7,16..23}
        v16h afrag[4];
        const float* arow = node_embeds + (size_t)(m0 + l16) * DMODEL;
        #pragma unroll
        for (int kc = 0; kc < 4; ++kc) {
            const int k0 = kc * 32 + half * 8;
            #pragma unroll
            for (int e = 0; e < 8; ++e) {
                afrag[kc][e]     = (_Float16)arow[k0 + e];
                afrag[kc][e + 8] = (_Float16)arow[k0 + 16 + e];
            }
        }

        #pragma unroll
        for (int t = 0; t < 2; ++t) {
            const int nt = wave * 2 + t;                 // N-tile 0..15
            const float* brow = hyper_embeds + (size_t)(nt * 16 + l16) * DMODEL;
            v8f acc = {};
            #pragma unroll
            for (int kc = 0; kc < 4; ++kc) {
                // B fragment (32x16), same K order as A: half==0 -> K {0..7,16..23}, half==1 -> K {8..15,24..31}
                v16h bfrag;
                const int k0 = kc * 32 + half * 8;
                #pragma unroll
                for (int e = 0; e < 8; ++e) {
                    bfrag[e]     = (_Float16)brow[k0 + e];
                    bfrag[e + 8] = (_Float16)brow[k0 + 16 + e];
                }
                acc = __builtin_amdgcn_wmma_f32_16x16x32_f16(
                        false, afrag[kc], false, bfrag, (short)0, acc, false, false);
            }
            // C/D layout: vgpr r: lanes<16 -> (M=r, N=l16); lanes>=16 -> (M=r+8, N=l16)
            #pragma unroll
            for (int r = 0; r < 8; ++r)
                simS[r + half * 8][nt * 16 + l16] = acc[r];
        }
    }

    __syncthreads();

    // ---- Phase 2: relu*penalty, softmax over 256, top-8 cut, threshold ----
    {
        const int row = tid >> 4;     // 16 lanes per row
        const int sub = tid & 15;     // columns sub + 16*j
        const float pr = pen[row];

        float p[16];
        float lmax = -__builtin_inff();
        #pragma unroll
        for (int j = 0; j < 16; ++j) {
            float v = simS[row][sub + 16 * j];
            v = fmaxf(v, 0.0f) * pr;          // ALPHA == 1.0
            p[j] = v;
            lmax = fmaxf(lmax, v);
        }
        #pragma unroll
        for (int off = 1; off < 16; off <<= 1)
            lmax = fmaxf(lmax, __shfl_xor(lmax, off, 32));

        float lsum = 0.0f;
        #pragma unroll
        for (int j = 0; j < 16; ++j) {
            p[j] = __expf(p[j] - lmax);
            lsum += p[j];
        }
        #pragma unroll
        for (int off = 1; off < 16; off <<= 1)
            lsum += __shfl_xor(lsum, off, 32);
        const float inv = 1.0f / lsum;
        #pragma unroll
        for (int j = 0; j < 16; ++j) p[j] *= inv;

        // 8th-largest via repeated strictly-descending max with tie counting
        float cut = __builtin_inff();
        int remaining = KTOP;
        for (int it = 0; it < KTOP && remaining > 0; ++it) {
            float lm = -1.0f;
            #pragma unroll
            for (int j = 0; j < 16; ++j)
                lm = (p[j] < cut) ? fmaxf(lm, p[j]) : lm;
            #pragma unroll
            for (int off = 1; off < 16; off <<= 1)
                lm = fmaxf(lm, __shfl_xor(lm, off, 32));
            int cnt = 0;
            #pragma unroll
            for (int j = 0; j < 16; ++j) cnt += (p[j] == lm) ? 1 : 0;
            #pragma unroll
            for (int off = 1; off < 16; off <<= 1)
                cnt += __shfl_xor(cnt, off, 32);
            cut = lm;
            remaining -= cnt;
        }

        #pragma unroll
        for (int j = 0; j < 16; ++j)
            simS[row][sub + 16 * j] = (p[j] >= cut && p[j] > THRESH) ? 1.0f : 0.0f;
    }

    __syncthreads();

    // ---- Phase 3: coalesced b128 non-temporal stores (16 KB per block) ----
    {
        float* obase = out + (size_t)m0 * HYPER;
        #pragma unroll
        for (int it = 0; it < 4; ++it) {
            const int i = tid * 4 + it * 1024;      // 0..4095
            const int r = i >> 8;
            const int c = i & 255;
            v4f v = *(const v4f*)&simS[r][c];       // rows are 16B-aligned (276*4)
            __builtin_nontemporal_store(v, (v4f*)&obase[i]);
        }
    }
}

extern "C" void kernel_launch(void* const* d_in, const int* in_sizes, int n_in,
                              void* d_out, int out_size, void* d_ws, size_t ws_size,
                              hipStream_t stream) {
    // inputs: 0=features (UNUSED by reference), 1=mask, 2=node_embeds, 3=hyper_embeds
    const float* mask  = (const float*)d_in[1];
    const float* node  = (const float*)d_in[2];
    const float* hyper = (const float*)d_in[3];
    float* out = (float*)d_out;

    (void)in_sizes; (void)n_in; (void)out_size; (void)d_ws; (void)ws_size;
    hypergraph_fused_kernel<<<SEQ / 16, 256, 0, stream>>>(mask, node, hyper, out);
}
